// AdapterAttentionBlock_14766097563926
// MI455X (gfx1250) — matrix operand order repaired and checked by reading, not compiled, not run on hardware
//
#include <hip/hip_runtime.h>
#include <hip/hip_bf16.h>
#include <math.h>

// ---------------------------------------------------------------------------
// AdapterAttentionBlock for MI455X (gfx1250, wave32, WMMA + TDM).
// f16 dataflow: all GEMM operands are f16 (f32 accum); residuals stay f32.
// GEMM: 128x128 block tile, K-step 32, 8 waves, 8 v_wmma per wave per step.
// Tile staging via Tensor Data Mover (double-buffered LDS) when available.
// ---------------------------------------------------------------------------

typedef __attribute__((ext_vector_type(16))) _Float16 v16h;
typedef __attribute__((ext_vector_type(8)))  _Float16 v8h;
typedef __attribute__((ext_vector_type(4)))  _Float16 v4h;
typedef __attribute__((ext_vector_type(2)))  _Float16 v2h;
typedef __attribute__((ext_vector_type(8)))  float    v8f;
typedef __attribute__((ext_vector_type(4)))  unsigned int v4u;
typedef __attribute__((ext_vector_type(8)))  int v8i;
typedef __attribute__((ext_vector_type(4)))  int v4i;

#define DMODEL 768
#define HEADS  12
#define HDIM   64
#define LSEQ   197
#define BTOK   32
#define NGRP   16
#define E3     2304
#define M0     (LSEQ * BTOK)   // 6304 token rows

#define BM 128
#define BN 128
#define BK 32
#define LDSK 40   // padded LDS row stride in halves (80B, 16B-aligned chunks)

#if defined(__has_builtin)
#if __has_builtin(__builtin_amdgcn_tensor_load_to_lds) && \
    __has_builtin(__builtin_amdgcn_s_wait_tensorcnt)
#define HAVE_TDM 1
#endif
#endif

// ----------------------------- TDM tile load -------------------------------
// Loads a (BK x 128)-element f16 tile (tile_dim0=32 along K, tile_dim1=128
// rows) from a row-major (rows x K) f16 tensor into LDS at lds_byte_off,
// inserting 4 dwords of padding after every 16 dwords -> LDS row stride 80B.
// Rows beyond rows_total read zero (handles the M edge).
__device__ __forceinline__ void tdm_load_tile(const _Float16* src, int K,
                                              int rows_total, long long row0,
                                              int k0, unsigned lds_byte_off) {
#ifdef HAVE_TDM
  unsigned long long ga =
      (unsigned long long)(uintptr_t)src + ((unsigned long long)(row0 * K + k0)) * 2ull;
  v4u g0 = {1u,                                   // count=1, user descriptor
            lds_byte_off,                         // lds_addr
            (unsigned)(ga & 0xffffffffu),         // global_addr[31:0]
            (unsigned)((ga >> 32) & 0x1ffffffu) | (2u << 30)};  // addr hi, type=2
  int w0 = (1 << 16)    // data_size = 2 bytes
         | (1 << 20)    // pad_enable
         | (3 << 22)    // pad_interval: 16 dwords
         | (3 << 25);   // pad_amount: 4 dwords
  v8i g1 = {w0,
            (int)((K & 0xffff) << 16),                               // dim0 lo
            (int)(((unsigned)K >> 16) | ((rows_total & 0xffff) << 16)),
            (int)(((unsigned)rows_total >> 16) | (32u << 16)),       // tile_dim0
            128,                                                     // tile_dim1
            K,                                                       // dim0 stride
            0, 0};
  v4i g2 = {0, 0, 0, 0};
  v4i g3 = {0, 0, 0, 0};
#if __clang_major__ >= 23
  v8i g4 = {0, 0, 0, 0, 0, 0, 0, 0};
  __builtin_amdgcn_tensor_load_to_lds(g0, g1, g2, g3, g4, 0);
#else
  __builtin_amdgcn_tensor_load_to_lds(g0, g1, g2, g3, 0);
#endif
#else
  (void)src; (void)K; (void)rows_total; (void)row0; (void)k0; (void)lds_byte_off;
#endif
}

// ------------------------------ LayerNorm ----------------------------------
// f32 in, f16 out (output is only ever consumed as a GEMM A operand).
__global__ __launch_bounds__(256) void ln_kernel(
    const float* __restrict__ x, const float* __restrict__ g,
    const float* __restrict__ b, _Float16* __restrict__ y, int M)
{
  __shared__ float red[256];
  int row = blockIdx.x;
  if (row >= M) return;
  int t = threadIdx.x;
  const float* xr = x + (size_t)row * DMODEL;
  float v0 = xr[t], v1 = xr[t + 256], v2 = xr[t + 512];
  red[t] = v0 + v1 + v2;
  __syncthreads();
  for (int off = 128; off > 0; off >>= 1) {
    if (t < off) red[t] += red[t + off];
    __syncthreads();
  }
  float mu = red[0] * (1.0f / 768.0f);
  __syncthreads();
  float d0 = v0 - mu, d1 = v1 - mu, d2 = v2 - mu;
  red[t] = d0 * d0 + d1 * d1 + d2 * d2;
  __syncthreads();
  for (int off = 128; off > 0; off >>= 1) {
    if (t < off) red[t] += red[t + off];
    __syncthreads();
  }
  float r = rsqrtf(red[0] * (1.0f / 768.0f) + 1e-5f);
  _Float16* yr = y + (size_t)row * DMODEL;
  yr[t]       = (_Float16)(d0 * r * g[t]       + b[t]);
  yr[t + 256] = (_Float16)(d1 * r * g[t + 256] + b[t + 256]);
  yr[t + 512] = (_Float16)(d2 * r * g[t + 512] + b[t + 512]);
}

// --------------------------- WMMA inner step -------------------------------
__device__ __forceinline__ void mma_step(const _Float16* As, const _Float16* Bs,
                                         int wm, int wn, int lm, int lk,
                                         v8f acc[4][2])
{
  // A frag (ISA 16-bit A 16x32): lane row lm; halves 0..7 = K lk*8..+7,
  // halves 8..15 = K 16+lk*8..+7.
  v16h aF[4];
#pragma unroll
  for (int i = 0; i < 4; ++i) {
    const _Float16* ap = As + (wm * 64 + i * 16 + lm) * LDSK + lk * 8;
    v8h lo = *(const v8h*)ap;
    v8h hi = *(const v8h*)(ap + 16);
    aF[i] = __builtin_shufflevector(lo, hi, 0, 1, 2, 3, 4, 5, 6, 7,
                                    8, 9, 10, 11, 12, 13, 14, 15);
  }
  // B frag (32x16): lane column lm; halves 0..7 = K lk*8..+7, halves 8..15 = K 16+lk*8..+7.
  v16h bF[2];
#pragma unroll
  for (int j = 0; j < 2; ++j) {
    const _Float16* bp = Bs + (wn * 32 + j * 16 + lm) * LDSK + lk * 8;
    v8h lo = *(const v8h*)bp;
    v8h hi = *(const v8h*)(bp + 16);
    bF[j] = __builtin_shufflevector(lo, hi, 0, 1, 2, 3, 4, 5, 6, 7,
                                    8, 9, 10, 11, 12, 13, 14, 15);
  }
#pragma unroll
  for (int i = 0; i < 4; ++i)
#pragma unroll
    for (int j = 0; j < 2; ++j)
      acc[i][j] = __builtin_amdgcn_wmma_f32_16x16x32_f16(
          false, aF[i], false, bF[j], (short)0, acc[i][j], false, false);
}

// ------------------------------ WMMA GEMM ----------------------------------
// C[M,N] = act(A[M,K] @ W[N,K]^T + bias[N]) (+ res[M,N])
// A, W: f16 row-major.  res: f32.  C: f16 (OUTH) or f32.
template <int ACT, bool RES, bool OUTH>
__global__ __launch_bounds__(256) void gemm_wmma_kernel(
    const _Float16* __restrict__ A, const _Float16* __restrict__ W,
    const float* __restrict__ bias, const float* __restrict__ res,
    void* __restrict__ Cout, int M, int N, int K)
{
#ifdef HAVE_TDM
#define NBUF 2
#else
#define NBUF 1
#endif
  __shared__ _Float16 As[NBUF][BM * LDSK];
  __shared__ _Float16 Bs[NBUF][BN * LDSK];

  int tid  = threadIdx.x;
  int lane = tid & 31;
  int wid  = tid >> 5;
  int wm   = wid & 1;        // 2 waves along M
  int wn   = wid >> 1;       // 4 waves along N
  int m0   = blockIdx.x * BM;
  int n0   = blockIdx.y * BN;
  int lm   = lane & 15;
  int lk   = lane >> 4;

  v8f acc[4][2];
#pragma unroll
  for (int i = 0; i < 4; ++i)
#pragma unroll
    for (int j = 0; j < 2; ++j)
#pragma unroll
      for (int r = 0; r < 8; ++r) acc[i][j][r] = 0.0f;

  const int S = K / BK;

#ifdef HAVE_TDM
  // ---- TDM double-buffered pipeline: wave 0 drives the DMA. ----
  if (tid < 32) {
    tdm_load_tile(A, K, M, m0, 0, (unsigned)(uintptr_t)&As[0][0]);
    tdm_load_tile(W, K, N, n0, 0, (unsigned)(uintptr_t)&Bs[0][0]);
  }
  for (int k = 0; k < S; ++k) {
    int buf = k & 1;
    if (tid < 32) {
      if (k + 1 < S) {
        int nb = buf ^ 1;
        tdm_load_tile(A, K, M, m0, (k + 1) * BK, (unsigned)(uintptr_t)&As[nb][0]);
        tdm_load_tile(W, K, N, n0, (k + 1) * BK, (unsigned)(uintptr_t)&Bs[nb][0]);
        __builtin_amdgcn_s_wait_tensorcnt(2);   // current buffer complete
      } else {
        __builtin_amdgcn_s_wait_tensorcnt(0);
      }
    }
    __syncthreads();
    mma_step(&As[buf][0], &Bs[buf][0], wm, wn, lm, lk, acc);
    __syncthreads();   // all reads done before this buffer is refilled
  }
#else
  // ---- Fallback: cooperative vector loads (pure f16, no conversion). ----
  for (int k = 0; k < S; ++k) {
    int k0 = k * BK;
#pragma unroll
    for (int j = 0; j < 2; ++j) {
      int i = j * 256 + tid;          // 0..511 chunks of 8 halves
      int r = i >> 2;                 // row 0..127
      int c = (i & 3) << 3;           // 0,8,16,24
      int gr = m0 + r; if (gr >= M) gr = M - 1;
      *(v8h*)&As[0][r * LDSK + c] = *(const v8h*)(A + (size_t)gr * K + k0 + c);
      *(v8h*)&Bs[0][r * LDSK + c] = *(const v8h*)(W + (size_t)(n0 + r) * K + k0 + c);
    }
    __syncthreads();
    mma_step(&As[0][0], &Bs[0][0], wm, wn, lm, lk, acc);
    __syncthreads();
  }
#endif

  // ---- Epilogue: C/D layout vgpr r -> M = r + lk*8, N = lm. ----
#pragma unroll
  for (int i = 0; i < 4; ++i) {
#pragma unroll
    for (int j = 0; j < 2; ++j) {
      int gn = n0 + wn * 32 + j * 16 + lm;
      float bv = bias[gn];
      int gmb = m0 + wm * 64 + i * 16 + lk * 8;
#pragma unroll
      for (int r = 0; r < 8; ++r) {
        int gm = gmb + r;
        if (gm < M) {
          float v = acc[i][j][r] + bv;
          if (ACT == 1) v = v * (1.0f / (1.0f + __expf(-1.702f * v)));
          if (RES) v += res[(size_t)gm * N + gn];
          if (OUTH) ((_Float16*)Cout)[(size_t)gm * N + gn] = (_Float16)v;
          else      ((float*)Cout)[(size_t)gm * N + gn] = v;
        }
      }
    }
  }
}

// ------------------------------ Attention ----------------------------------
// One wave per query row; qkv rows [(l*32+b), 2304] f16 with [q|k|v] slices.
// mask != null => cross attention (additive 0/1 mask, implicit CLS 1).
__global__ __launch_bounds__(128) void attn_kernel(
    const _Float16* __restrict__ qkv_q, const _Float16* __restrict__ qkv_kv,
    const float* __restrict__ mask, _Float16* __restrict__ out, int Lq)
{
  const int Lk = LSEQ;
  int wid  = threadIdx.x >> 5;
  int lane = threadIdx.x & 31;
  int lq   = blockIdx.x * 4 + wid;
  int bh   = blockIdx.y;
  int b = bh / HEADS, h = bh % HEADS;
  if (lq >= Lq) return;

  const _Float16* qrow = qkv_q + (size_t)(lq * BTOK + b) * E3 + h * HDIM;
  float q0 = (float)qrow[lane * 2], q1 = (float)qrow[lane * 2 + 1];
  const float scale = 0.125f;    // 1/sqrt(64)

  float s[7];
#pragma unroll
  for (int i = 0; i < 7; ++i) {
    int cnt = Lk - i * 32; if (cnt > 32) cnt = 32;
    float sv = -3.0e38f;
    for (int j = 0; j < cnt; ++j) {
      int l = i * 32 + j;
      const _Float16* kr =
          qkv_kv + (size_t)(l * BTOK + b) * E3 + DMODEL + h * HDIM + lane * 2;
      v2h kv = *(const v2h*)kr;
      float p = q0 * (float)kv[0] + q1 * (float)kv[1];
#pragma unroll
      for (int off = 16; off > 0; off >>= 1) p += __shfl_xor(p, off, 32);
      p *= scale;
      if (mask) {
        float mv = (l == 0) ? 1.0f
                            : mask[((size_t)b * NGRP + lq) * (LSEQ - 1) + (l - 1)];
        p += mv;
      }
      if (j == lane) sv = p;
    }
    s[i] = sv;
  }

  float mx = s[0];
#pragma unroll
  for (int i = 1; i < 7; ++i) mx = fmaxf(mx, s[i]);
#pragma unroll
  for (int off = 16; off > 0; off >>= 1) mx = fmaxf(mx, __shfl_xor(mx, off, 32));

  float sum = 0.0f;
#pragma unroll
  for (int i = 0; i < 7; ++i) {
    float e = (i * 32 + lane < Lk) ? __expf(s[i] - mx) : 0.0f;
    s[i] = e;
    sum += e;
  }
#pragma unroll
  for (int off = 16; off > 0; off >>= 1) sum += __shfl_xor(sum, off, 32);
  float inv = 1.0f / sum;

  float o0 = 0.0f, o1 = 0.0f;
#pragma unroll
  for (int i = 0; i < 7; ++i) {
    int cnt = Lk - i * 32; if (cnt > 32) cnt = 32;
    for (int j = 0; j < cnt; ++j) {
      int l = i * 32 + j;
      float a = __shfl(s[i], j, 32) * inv;
      const _Float16* vr =
          qkv_kv + (size_t)(l * BTOK + b) * E3 + 2 * DMODEL + h * HDIM + lane * 2;
      v2h vv = *(const v2h*)vr;
      o0 += a * (float)vv[0];
      o1 += a * (float)vv[1];
    }
  }
  _Float16* orow = out + (size_t)(lq * BTOK + b) * DMODEL + h * HDIM + lane * 2;
  orow[0] = (_Float16)o0;
  orow[1] = (_Float16)o1;
}

// --------------------------- Pooling / misc --------------------------------
__global__ void cvt_kernel(const float* __restrict__ src,
                           _Float16* __restrict__ dst, int n)
{
  int i = (blockIdx.x * 256 + threadIdx.x) * 4;
  if (i < n) {
    float4 f = *(const float4*)(src + i);
    v4h h = {(_Float16)f.x, (_Float16)f.y, (_Float16)f.z, (_Float16)f.w};
    *(v4h*)(dst + i) = h;
  }
}

__global__ void presence_kernel(const float* __restrict__ mask,
                                float* __restrict__ present,
                                float* __restrict__ counts)
{
  int n = blockIdx.x;      // 16
  int b = threadIdx.x;     // 32
  const float* mr = mask + ((size_t)b * NGRP + n) * (LSEQ - 1);
  float srow = 0.0f;
  for (int l = 0; l < LSEQ - 1; ++l) srow += mr[l];
  float p = (srow > 0.0f) ? 1.0f : 0.0f;
  present[b * NGRP + n] = p;
  float t = srow + p;
#pragma unroll
  for (int off = 16; off > 0; off >>= 1) t += __shfl_xor(t, off, 32);
  if (b == 0) counts[n] = t;
}

__global__ __launch_bounds__(128) void feat_kernel(
    const float* __restrict__ xt, const float* __restrict__ mask,
    const float* __restrict__ present, const float* __restrict__ counts,
    float* __restrict__ feat)
{
  int n = blockIdx.x;
  int d = blockIdx.y * 128 + threadIdx.x;
  float acc = 0.0f;
  for (int b = 0; b < BTOK; ++b) {
    acc += present[b * NGRP + n] * xt[(size_t)b * DMODEL + d];  // CLS (l=0)
    const float* mr = mask + ((size_t)b * NGRP + n) * (LSEQ - 1);
    for (int l = 0; l < LSEQ - 1; ++l) {
      float m = mr[l];
      if (m != 0.0f)
        acc += m * xt[((size_t)(l + 1) * BTOK + b) * DMODEL + d];
    }
  }
  feat[n * DMODEL + d] = acc / counts[n];
}

__global__ void ybcast_kernel(const float* __restrict__ feat,
                              float* __restrict__ y)
{
  int i = blockIdx.x * 256 + threadIdx.x;     // 512*768 total
  int d = i % DMODEL;
  int n = (i / DMODEL) / BTOK;
  y[i] = feat[n * DMODEL + d];
}

__global__ __launch_bounds__(128) void yout_kernel(
    const float* __restrict__ feat, const float* __restrict__ present,
    float* __restrict__ yout)
{
  int b_ = blockIdx.x;                        // 4
  int d  = blockIdx.y * 128 + threadIdx.x;    // 768
  float num = 0.0f, den = 0.0f;
  for (int t = 0; t < 8; ++t)
    for (int n = 0; n < NGRP; ++n) {
      float p = present[(b_ * 8 + t) * NGRP + n];
      num += p * feat[n * DMODEL + d];
      den += p;
    }
  float v = num / den;
  for (int t = 0; t < 8; ++t)
    yout[(size_t)(b_ * 8 + t) * DMODEL + d] = v;
}

// ------------------------------- Host side ---------------------------------
struct BlkP {
  const float *in_w, *in_b, *out_w, *out_b, *ln1_g, *ln1_b,
              *ln2_g, *ln2_b, *fc_w, *fc_b, *proj_w, *proj_b;
};
struct HBlk { _Float16 *in_w, *out_w, *fc_w, *proj_w; };

static BlkP getBlk(void* const* d_in, int base) {
  BlkP p;
  p.in_w  = (const float*)d_in[base + 0];  p.in_b  = (const float*)d_in[base + 1];
  p.out_w = (const float*)d_in[base + 2];  p.out_b = (const float*)d_in[base + 3];
  p.ln1_g = (const float*)d_in[base + 4];  p.ln1_b = (const float*)d_in[base + 5];
  p.ln2_g = (const float*)d_in[base + 6];  p.ln2_b = (const float*)d_in[base + 7];
  p.fc_w  = (const float*)d_in[base + 8];  p.fc_b  = (const float*)d_in[base + 9];
  p.proj_w = (const float*)d_in[base + 10]; p.proj_b = (const float*)d_in[base + 11];
  return p;
}

static inline void cvt(const float* s, _Float16* d, int n, hipStream_t st) {
  cvt_kernel<<<(n / 4 + 255) / 256, 256, 0, st>>>(s, d, n);
}

// act/out dispatch: {f16 out, no res} x {act 0/1}, {f32 out + res}.
static inline void gemm_h(const _Float16* A, const _Float16* W,
                          const float* bias, _Float16* C, int M, int N, int K,
                          int act, hipStream_t s) {
  dim3 g((M + BM - 1) / BM, N / BN);
  if (act == 1)
    gemm_wmma_kernel<1, false, true><<<g, 256, 0, s>>>(A, W, bias, nullptr, C, M, N, K);
  else
    gemm_wmma_kernel<0, false, true><<<g, 256, 0, s>>>(A, W, bias, nullptr, C, M, N, K);
}
static inline void gemm_f(const _Float16* A, const _Float16* W,
                          const float* bias, const float* res, float* C,
                          int M, int N, int K, hipStream_t s) {
  dim3 g((M + BM - 1) / BM, N / BN);
  gemm_wmma_kernel<0, true, false><<<g, 256, 0, s>>>(A, W, bias, res, C, M, N, K);
}

static void block_fwd(const float* x, const BlkP& p, const HBlk& hw, float* out,
                      _Float16* hA, _Float16* hQKV, _Float16* hFC, float* bufC,
                      hipStream_t s) {
  ln_kernel<<<M0, 256, 0, s>>>(x, p.ln1_g, p.ln1_b, hA, M0);
  gemm_h(hA, hw.in_w, p.in_b, hQKV, M0, E3, DMODEL, 0, s);           // qkv (f16)
  attn_kernel<<<dim3((LSEQ + 3) / 4, BTOK * HEADS), 128, 0, s>>>(
      hQKV, hQKV, nullptr, hA, LSEQ);                                // o (f16)
  gemm_f(hA, hw.out_w, p.out_b, x, bufC, M0, DMODEL, DMODEL, s);     // x' (f32)
  ln_kernel<<<M0, 256, 0, s>>>(bufC, p.ln2_g, p.ln2_b, hA, M0);
  gemm_h(hA, hw.fc_w, p.fc_b, hFC, M0, 4 * DMODEL, DMODEL, 1, s);    // qgelu f16
  gemm_f(hFC, hw.proj_w, p.proj_b, bufC, out, M0, DMODEL, 4 * DMODEL, s);
}

extern "C" void kernel_launch(void* const* d_in, const int* in_sizes, int n_in,
                              void* d_out, int out_size, void* d_ws,
                              size_t ws_size, hipStream_t stream) {
  const float* X    = (const float*)d_in[0];
  const float* mask = (const float*)d_in[1];
  BlkP blk  = getBlk(d_in, 2);
  BlkP adp  = getBlk(d_in, 14);
  BlkP adp1 = getBlk(d_in, 26);

  float* out_xt1 = (float*)d_out;                            // (16, 32, 768)
  float* out_y   = out_xt1 + (size_t)NGRP * BTOK * DMODEL;   // (1, 32, 768)
  float* out_xt2 = out_y + (size_t)BTOK * DMODEL;            // (197, 32, 768)

  char* p = (char*)d_ws;
  auto alloc = [&](size_t bytes) {
    char* r = p; p += (bytes + 255) & ~(size_t)255; return r;
  };
  float* bufC   = (float*)alloc((size_t)M0 * DMODEL * 4);
  float* x12    = (float*)alloc((size_t)M0 * DMODEL * 4);
  float* x_temp = (float*)alloc((size_t)M0 * DMODEL * 4);
  float* ybuf   = (float*)alloc((size_t)NGRP * BTOK * DMODEL * 4);
  float* feat    = (float*)alloc(NGRP * DMODEL * 4);
  float* present = (float*)alloc(BTOK * NGRP * 4);
  float* counts  = (float*)alloc(NGRP * 4);
  _Float16* hA   = (_Float16*)alloc((size_t)M0 * DMODEL * 2);
  _Float16* hQKV = (_Float16*)alloc((size_t)M0 * E3 * 2);
  _Float16* hFC  = (_Float16*)alloc((size_t)M0 * 4 * DMODEL * 2);
  _Float16* lnY   = (_Float16*)alloc((size_t)NGRP * BTOK * DMODEL * 2);
  _Float16* qkvY  = (_Float16*)alloc((size_t)NGRP * BTOK * E3 * 2);
  _Float16* attnY = (_Float16*)alloc((size_t)NGRP * BTOK * DMODEL * 2);
  _Float16* lnT   = (_Float16*)alloc((size_t)NGRP * BTOK * DMODEL * 2);
  _Float16* fcY   = (_Float16*)alloc((size_t)NGRP * BTOK * 4 * DMODEL * 2);
  BlkP    blks[3] = {blk, adp, adp1};
  HBlk    hw[3];
  for (int i = 0; i < 3; ++i) {
    hw[i].in_w   = (_Float16*)alloc((size_t)E3 * DMODEL * 2);
    hw[i].out_w  = (_Float16*)alloc((size_t)DMODEL * DMODEL * 2);
    hw[i].fc_w   = (_Float16*)alloc((size_t)4 * DMODEL * DMODEL * 2);
    hw[i].proj_w = (_Float16*)alloc((size_t)DMODEL * 4 * DMODEL * 2);
  }
  (void)ws_size; (void)n_in; (void)in_sizes; (void)out_size;

  // One-time (per launch) f32 -> f16 weight conversion.
  for (int i = 0; i < 3; ++i) {
    cvt(blks[i].in_w,   hw[i].in_w,   E3 * DMODEL, stream);
    cvt(blks[i].out_w,  hw[i].out_w,  DMODEL * DMODEL, stream);
    cvt(blks[i].fc_w,   hw[i].fc_w,   4 * DMODEL * DMODEL, stream);
    cvt(blks[i].proj_w, hw[i].proj_w, DMODEL * 4 * DMODEL, stream);
  }

  // Three transformer blocks.
  block_fwd(X,   blk,  hw[0], x12,     hA, hQKV, hFC, bufC, stream);
  block_fwd(x12, adp,  hw[1], x_temp,  hA, hQKV, hFC, bufC, stream);
  block_fwd(x12, adp1, hw[2], out_xt2, hA, hQKV, hFC, bufC, stream);  // x_temp2

  // Masked group pooling -> feat (16,768), y (16,32,768).
  presence_kernel<<<NGRP, 32, 0, stream>>>(mask, present, counts);
  feat_kernel<<<dim3(NGRP, DMODEL / 128), 128, 0, stream>>>(
      x_temp, mask, present, counts, feat);
  ybcast_kernel<<<(NGRP * BTOK * DMODEL) / 256, 256, 0, stream>>>(feat, ybuf);

  // Cross attention: q from ln(y), k/v from ln(x), blk params + mask.
  ln_kernel<<<M0, 256, 0, stream>>>(X, blk.ln1_g, blk.ln1_b, hA, M0);
  gemm_h(hA, hw[0].in_w, blk.in_b, hQKV, M0, E3, DMODEL, 0, stream);
  ln_kernel<<<NGRP * BTOK, 256, 0, stream>>>(ybuf, blk.ln1_g, blk.ln1_b, lnY,
                                             NGRP * BTOK);
  gemm_h(lnY, hw[0].in_w, blk.in_b, qkvY, NGRP * BTOK, E3, DMODEL, 0, stream);
  attn_kernel<<<dim3((NGRP + 3) / 4, BTOK * HEADS), 128, 0, stream>>>(
      qkvY, hQKV, mask, attnY, NGRP);
  gemm_f(attnY, hw[0].out_w, blk.out_b, ybuf, out_xt1, NGRP * BTOK, DMODEL,
         DMODEL, stream);                                   // xt1 = mha + y
  ln_kernel<<<NGRP * BTOK, 256, 0, stream>>>(out_xt1, blk.ln2_g, blk.ln2_b,
                                             lnT, NGRP * BTOK);
  gemm_h(lnT, hw[0].fc_w, blk.fc_b, fcY, NGRP * BTOK, 4 * DMODEL, DMODEL, 1,
         stream);
  gemm_f(fcY, hw[0].proj_w, blk.proj_b, out_xt1, out_xt1, NGRP * BTOK, DMODEL,
         4 * DMODEL, stream);                               // in-place residual

  // y_out: masked mean of feat per batch-of-T, repeated T times.
  yout_kernel<<<dim3(BTOK / 8, DMODEL / 128), 128, 0, stream>>>(feat, present,
                                                                out_y);
}
